// BaseKernelSetConv_21689584845341
// MI455X (gfx1250) — matrix operand order repaired and checked by reading, not compiled, not run on hardware
//
#include <hip/hip_runtime.h>

// CDNA5 (gfx1250) WMMA / vector types
typedef __attribute__((ext_vector_type(16))) _Float16 v16h;
typedef __attribute__((ext_vector_type(8)))  float    v8f;
typedef __attribute__((ext_vector_type(4)))  float    v4f;

#define EPSN 1e-8f

__device__ __forceinline__ float rcp_fast(float x) {
  return __builtin_amdgcn_rcpf(x);   // v_rcp_f32, 1-ulp
}

// Load one 32-float row of a (16 x 32) A-matrix, L2-normalize it across the
// lane pair (lane ^ 16), scale, convert to f16 in the contiguous k order used here (same as unit_vecB):
//   halves 0..7  : K = 16*hi + 0..7
//   halves 8..15 : K = 16*hi + 8..15
__device__ __forceinline__ v16h unit_rowA(const float* __restrict__ row,
                                          int hi, float scale) {
  float a[16];
  float ss = 0.f;
#pragma unroll
  for (int i = 0; i < 8; ++i) { float v = row[16 * hi + i];     a[i]     = v; ss += v * v; }
#pragma unroll
  for (int i = 0; i < 8; ++i) { float v = row[16 * hi + 8 + i]; a[8 + i] = v; ss += v * v; }
  ss += __shfl_xor(ss, 16);                 // lane pair covers all 32 features
  const float inv = scale * rcp_fast(sqrtf(ss) + EPSN);
  v16h r;
#pragma unroll
  for (int i = 0; i < 16; ++i) r[i] = (_Float16)(a[i] * inv);
  return r;
}

// Load one node feature vector (32 floats) as the WMMA B-operand column:
// lane pair (n, n+16) holds features [0..15] / [16..31]; normalize jointly.
__device__ __forceinline__ v16h unit_vecB(const float* __restrict__ xrow, int hi) {
  const v4f* p = (const v4f*)xrow + hi * 4;   // 64B-aligned
  v4f f[4];
#pragma unroll
  for (int q = 0; q < 4; ++q) f[q] = p[q];
  float ss = 0.f;
#pragma unroll
  for (int q = 0; q < 4; ++q)
    ss += f[q][0] * f[q][0] + f[q][1] * f[q][1] +
          f[q][2] * f[q][2] + f[q][3] * f[q][3];
  ss += __shfl_xor(ss, 16);
  const float inv = rcp_fast(sqrtf(ss) + EPSN);
  v16h r;
#pragma unroll
  for (int q = 0; q < 4; ++q) {
    r[4 * q + 0] = (_Float16)(f[q][0] * inv);
    r[4 * q + 1] = (_Float16)(f[q][1] * inv);
    r[4 * q + 2] = (_Float16)(f[q][2] * inv);
    r[4 * q + 3] = (_Float16)(f[q][3] * inv);
  }
  return r;
}

// One wave computes TILES tiles of 16 nodes each. Per tile: a 16(kernels) x
// 16(nodes) score block via (1 + DEG) chained v_wmma_f32_16x16x32_f16, then
// each lane writes its own 8 contiguous scores (k = 8*hi..8*hi+7 live at
// row offset CS*16 + 8*hi -- no cross-lane exchange needed) plus 6 of the 12
// zero float4 slots. All output stores are non-temporal to keep x resident
// in the 192MB L2.
template <int DEG, int TILES>
__global__ __launch_bounds__(128) void setconv_score(
    const float* __restrict__ x,    // (N, 32)
    const int*   __restrict__ sel,  // (Nd)
    const int*   __restrict__ nei,  // (Nd * DEG)
    const float* __restrict__ Wf,   // (16, 32)
    const float* __restrict__ Wn,   // (16, DEG, 32)
    float*       __restrict__ out,  // (N, 64)
    int Nd)
{
  const int lane = threadIdx.x & 31;
  const int wave = threadIdx.x >> 5;
  const int n    = lane & 15;   // node-in-tile / A-row
  const int hi   = lane >> 4;   // half selector

  const int base0 = (blockIdx.x * 4 + wave) * (16 * TILES);
  if (base0 >= Nd) return;      // wave-uniform exit: EXEC stays all-ones below

  // A operands built once per wave: unit(Wf), unit(Wn[:,j,:]) * (1/DEG).
  const v16h Af = unit_rowA(Wf + n * 32, hi, 1.0f);
  v16h An[DEG];
#pragma unroll
  for (int j = 0; j < DEG; ++j)
    An[j] = unit_rowA(Wn + n * (DEG * 32) + j * 32, hi, 1.0f / (float)DEG);

#pragma unroll 1
  for (int t = 0; t < TILES; ++t) {
    const int base = base0 + t * 16;
    if (base >= Nd) break;      // wave-uniform

    // Tail: clamp (redundant compute), predicate only the stores.
    const int i    = (base + n < Nd) ? (base + n) : (Nd - 1);
    const int node = sel[i];

    v8f acc = {};
    {
      const v16h Bf = unit_vecB(x + (size_t)node * 32, hi);
      acc = __builtin_amdgcn_wmma_f32_16x16x32_f16(false, Af, false, Bf,
                                                   (short)0, acc, false, false);
    }
#pragma unroll
    for (int j = 0; j < DEG; ++j) {
      const int  nb = nei[i * DEG + j];
      const v16h Bn = unit_vecB(x + (size_t)nb * 32, hi);
      acc = __builtin_amdgcn_wmma_f32_16x16x32_f16(false, An[j], false, Bn,
                                                   (short)0, acc, false, false);
    }

    if (base + n < Nd) {
      constexpr int CS = DEG - 1;                 // chunk holding the scores
      float* row = out + (size_t)node * 64;

      // My 8 scores are contiguous: row[CS*16 + 8*hi .. +7].
      v4f s0 = {acc[0], acc[1], acc[2], acc[3]};
      v4f s1 = {acc[4], acc[5], acc[6], acc[7]};
      v4f* ps = (v4f*)(row + CS * 16 + 8 * hi);   // 32B-aligned
      __builtin_nontemporal_store(s0, ps);
      __builtin_nontemporal_store(s1, ps + 1);

      // 12 zero float4 slots (chunks != CS), 6 per lane. Slot t maps to
      // float offset t*4, skipping the score chunk.
      const v4f z = {0.f, 0.f, 0.f, 0.f};
#pragma unroll
      for (int q = 0; q < 6; ++q) {
        const int tt  = 6 * hi + q;
        const int off = tt * 4 + ((tt >= 4 * CS) ? 16 : 0);
        __builtin_nontemporal_store(z, (v4f*)(row + off));
      }
    }
  }
}

extern "C" void kernel_launch(void* const* d_in, const int* in_sizes, int n_in,
                              void* d_out, int out_size, void* d_ws, size_t ws_size,
                              hipStream_t stream) {
  (void)n_in; (void)out_size; (void)d_ws; (void)ws_size;
  const float* x   = (const float*)d_in[0];
  float*       out = (float*)d_out;

  constexpr int TILES = 4;                 // nodes per wave = 16*TILES = 64
  constexpr int NODES_PER_BLOCK = 16 * TILES * 4;  // 4 waves / block

  // setup_inputs() dict order: x, then per degree d: sel, nei, W_focal, W_nei
#define LAUNCH_DEG(D, B)                                                       \
  do {                                                                         \
    const int*   sel = (const int*)d_in[(B) + 0];                              \
    const int*   nei = (const int*)d_in[(B) + 1];                              \
    const float* Wf  = (const float*)d_in[(B) + 2];                            \
    const float* Wn  = (const float*)d_in[(B) + 3];                            \
    const int    Nd  = in_sizes[(B) + 0];                                      \
    const int    blocks = (Nd + NODES_PER_BLOCK - 1) / NODES_PER_BLOCK;        \
    setconv_score<D, TILES><<<blocks, 128, 0, stream>>>(x, sel, nei, Wf, Wn,   \
                                                        out, Nd);              \
  } while (0)

  LAUNCH_DEG(1, 1);
  LAUNCH_DEG(2, 5);
  LAUNCH_DEG(3, 9);
  LAUNCH_DEG(4, 13);
#undef LAUNCH_DEG
}
